// GraphMambaLayer_38422777430224
// MI455X (gfx1250) — compile-verified, matrix operand order checked by reading, not run on hardware
//
#include <hip/hip_runtime.h>
#include <hip/hip_bf16.h>
#include <cstddef>
#include <cstdint>

// ---------------- static config (mirrors reference) ----------------
constexpr int NG     = 32;
constexpr int LSEQ   = 1024;
constexpr int NTOT   = NG * LSEQ;          // 32768
constexpr int EEDGE  = 16 * NTOT;          // 524288
constexpr int DIN    = 128;
constexpr int DMODEL = 256;
constexpr int DINNER = 512;
constexpr int DSTATE = 16;
constexpr int DTRANK = 16;
constexpr int DXDBL  = DTRANK + 2 * DSTATE; // 48

typedef _Float16 v4h  __attribute__((ext_vector_type(4)));
typedef _Float16 v8h  __attribute__((ext_vector_type(8)));
typedef _Float16 v16h __attribute__((ext_vector_type(16)));
typedef float    v8f  __attribute__((ext_vector_type(8)));

// =====================================================================
// Generic WMMA GEMM:  C[M,Nc] = A[M,K] * B[Nc,K]^T (+ bias[n])
// Block tile 128x64, 256 threads = 8 waves, each wave owns a 32x32 C tile
// (4 x v_wmma_f32_16x16x32_f16 per K-step). fp32->fp16 conversion on the
// LDS staging path, fp32 accumulation.
// Requires: M % 128 == 0, Nc % 64 == 0, K % 32 == 0.
// =====================================================================
__global__ __launch_bounds__(256) void wmma_gemm_kernel(
    const float* __restrict__ A, const float* __restrict__ B,
    const float* __restrict__ bias, float* __restrict__ C,
    int M, int Nc, int K)
{
    constexpr int TM = 128, TN = 64, TK = 32, LDS_K = TK + 8; // pad keeps 16B align
    __shared__ __align__(16) _Float16 As[TM][LDS_K];
    __shared__ __align__(16) _Float16 Bs[TN][LDS_K];

    const int tid   = threadIdx.x;
    const int lane  = tid & 31;
    const int wave  = tid >> 5;      // 0..7
    const int wm    = wave >> 1;     // 0..3  (M sub-tile group)
    const int wn    = wave & 1;      // 0..1  (N sub-tile group)
    const int m0    = blockIdx.y * TM;
    const int n0    = blockIdx.x * TN;
    const int row16 = lane & 15;
    const int koff  = (lane < 16) ? 0 : 8;   // ISA K-chunk select per half-wave

    v8f acc[2][2];
    const v8f vzero = {0.f,0.f,0.f,0.f,0.f,0.f,0.f,0.f};
    acc[0][0] = vzero; acc[0][1] = vzero; acc[1][0] = vzero; acc[1][1] = vzero;

    for (int k0 = 0; k0 < K; k0 += TK) {
        // stage A tile: 128x32 f32 -> f16   (1024 float4 loads / 256 thr = 4 each)
        #pragma unroll
        for (int i = 0; i < 4; ++i) {
            int v  = tid + i * 256;          // 0..1023
            int r  = v >> 3;                 // 8 float4 per 32-wide row
            int c4 = (v & 7) << 2;
            float4 f = *(const float4*)(A + (size_t)(m0 + r) * K + k0 + c4);
            v4h hv = {(_Float16)f.x, (_Float16)f.y, (_Float16)f.z, (_Float16)f.w};
            *(v4h*)&As[r][c4] = hv;
        }
        // stage B tile: 64x32
        #pragma unroll
        for (int i = 0; i < 2; ++i) {
            int v  = tid + i * 256;
            int r  = v >> 3;
            int c4 = (v & 7) << 2;
            float4 f = *(const float4*)(B + (size_t)(n0 + r) * K + k0 + c4);
            v4h hv = {(_Float16)f.x, (_Float16)f.y, (_Float16)f.z, (_Float16)f.w};
            *(v4h*)&Bs[r][c4] = hv;
        }
        __syncthreads();

        v16h afrag[2], bfrag[2];
        #pragma unroll
        for (int mi = 0; mi < 2; ++mi) {
            int r = wm * 32 + mi * 16 + row16;
            v8h lo = *(const v8h*)&As[r][koff];
            v8h hi = *(const v8h*)&As[r][koff + 16];
            afrag[mi] = __builtin_shufflevector(lo, hi, 0,1,2,3,4,5,6,7,8,9,10,11,12,13,14,15);
        }
        #pragma unroll
        for (int ni = 0; ni < 2; ++ni) {
            int r = wn * 32 + ni * 16 + row16;
            v8h lo = *(const v8h*)&Bs[r][koff];
            v8h hi = *(const v8h*)&Bs[r][koff + 16];
            bfrag[ni] = __builtin_shufflevector(lo, hi, 0,1,2,3,4,5,6,7,8,9,10,11,12,13,14,15);
        }
        #pragma unroll
        for (int mi = 0; mi < 2; ++mi)
            #pragma unroll
            for (int ni = 0; ni < 2; ++ni)
                acc[mi][ni] = __builtin_amdgcn_wmma_f32_16x16x32_f16(
                    false, afrag[mi], false, bfrag[ni],
                    (short)0, acc[mi][ni], false, false);
        __syncthreads();
    }

    // epilogue: C/D layout — lane&15 = N, VGPR r = M=r (+8 for upper half-wave)
    const int mhi = (lane < 16) ? 0 : 8;
    #pragma unroll
    for (int mi = 0; mi < 2; ++mi)
        #pragma unroll
        for (int ni = 0; ni < 2; ++ni) {
            int n = n0 + wn * 32 + ni * 16 + row16;
            float bv = bias ? bias[n] : 0.0f;
            #pragma unroll
            for (int r = 0; r < 8; ++r) {
                int m = m0 + wm * 32 + mi * 16 + r + mhi;
                C[(size_t)m * Nc + n] = acc[mi][ni][r] + bv;
            }
        }
}

// =====================================================================
// degree count
// =====================================================================
__global__ void zero_u32_kernel(unsigned* __restrict__ p, int n) {
    int i = blockIdx.x * blockDim.x + threadIdx.x;
    if (i < n) p[i] = 0u;
}
__global__ void degree_kernel(const int* __restrict__ src, unsigned* __restrict__ deg, int e) {
    int i = blockIdx.x * blockDim.x + threadIdx.x;
    if (i < e) {
        int s = src[i];
        atomicAdd(&deg[s], 1u);
    }
}

// =====================================================================
// per-graph stable argsort by descending degree (bitonic over packed keys)
// key = (deg << 11) | (1023 - idx); sort descending  ==> argsort(-deg) stable
// =====================================================================
__global__ __launch_bounds__(1024) void sort_kernel(
    const unsigned* __restrict__ deg, int* __restrict__ order)
{
    __shared__ unsigned keys[LSEQ];
    const int g = blockIdx.x, tid = threadIdx.x;
    keys[tid] = (deg[g * LSEQ + tid] << 11) | (unsigned)(LSEQ - 1 - tid);
    __syncthreads();
    for (int k = 2; k <= LSEQ; k <<= 1) {
        for (int j = k >> 1; j > 0; j >>= 1) {
            int ixj = tid ^ j;
            if (ixj > tid) {
                unsigned a = keys[tid], b = keys[ixj];
                bool up = ((tid & k) == 0);           // descending overall
                if (up ? (a < b) : (a > b)) { keys[tid] = b; keys[ixj] = a; }
            }
            __syncthreads();
        }
    }
    order[g * LSEQ + tid] = (LSEQ - 1) - (int)(keys[tid] & 0x7FFu);
}

// =====================================================================
// gather / scatter 256-float rows via permutation (float4)
// =====================================================================
__global__ void gather_kernel(const float* __restrict__ h, const int* __restrict__ order,
                              float* __restrict__ seq)
{
    int idx = blockIdx.x * blockDim.x + threadIdx.x;   // N * 64
    if (idx >= NTOT * (DMODEL / 4)) return;
    int row = idx >> 6, c4 = idx & 63;
    int g = row >> 10;
    int src = g * LSEQ + order[row];
    ((float4*)seq)[(size_t)row * 64 + c4] = ((const float4*)h)[(size_t)src * 64 + c4];
}
__global__ void scatter_kernel(const float* __restrict__ y, const int* __restrict__ order,
                               float* __restrict__ out)
{
    int idx = blockIdx.x * blockDim.x + threadIdx.x;
    if (idx >= NTOT * (DMODEL / 4)) return;
    int row = idx >> 6, c4 = idx & 63;
    int g = row >> 10;
    int dst = g * LSEQ + order[row];
    ((float4*)out)[(size_t)dst * 64 + c4] = ((const float4*)y)[(size_t)row * 64 + c4];
}

// =====================================================================
// causal depthwise conv1d (K=4) + SiLU ;  xi = xz[:, :512]
// =====================================================================
__global__ void conv_silu_kernel(const float* __restrict__ xz, const float* __restrict__ cw,
                                 const float* __restrict__ cb, float* __restrict__ u)
{
    int idx = blockIdx.x * blockDim.x + threadIdx.x;   // N * 512
    if (idx >= NTOT * DINNER) return;
    int row = idx >> 9, d = idx & 511;
    int l = row & (LSEQ - 1);
    const float* w = cw + d * 4;
    float s = cb[d];
    #pragma unroll
    for (int k = 0; k < 4; ++k) {
        int ls = l - 3 + k;
        if (ls >= 0) s += w[k] * xz[(size_t)(row - 3 + k) * (2 * DINNER) + d];
    }
    float sig = 1.0f / (1.0f + __expf(-s));
    u[idx] = s * sig;
}

// =====================================================================
// x_dbl = u @ x_proj_w^T   ([N,512] x [48,512]^T -> [N,48])
// =====================================================================
__global__ void xproj_kernel(const float* __restrict__ u, const float* __restrict__ W,
                             float* __restrict__ xdbl)
{
    int idx = blockIdx.x * blockDim.x + threadIdx.x;   // N * 48
    if (idx >= NTOT * DXDBL) return;
    int n = idx / DXDBL, j = idx - n * DXDBL;
    const float4* ur = (const float4*)(u + (size_t)n * DINNER);
    const float4* wr = (const float4*)(W + (size_t)j * DINNER);
    float s = 0.0f;
    #pragma unroll 4
    for (int i = 0; i < DINNER / 4; ++i) {
        float4 a = ur[i], b = wr[i];
        s += a.x * b.x + a.y * b.y + a.z * b.z + a.w * b.w;
    }
    xdbl[idx] = s;
}

// =====================================================================
// delta = softplus(dt @ dt_proj_w^T + dt_proj_b)   (K = 16)
// =====================================================================
__global__ void delta_kernel(const float* __restrict__ xdbl, const float* __restrict__ dtw,
                             const float* __restrict__ dtb, float* __restrict__ delta)
{
    int idx = blockIdx.x * blockDim.x + threadIdx.x;   // N * 512
    if (idx >= NTOT * DINNER) return;
    int n = idx >> 9, d = idx & 511;
    const float* xr = xdbl + (size_t)n * DXDBL;        // dt = first 16 cols
    const float* wr = dtw + d * DTRANK;
    float s = dtb[d];
    #pragma unroll
    for (int r = 0; r < DTRANK; ++r) s += xr[r] * wr[r];
    delta[idx] = (s > 20.0f) ? s : log1pf(__expf(s));
}

// =====================================================================
// selective scan + SiLU(z) gating, in-place over u.
// One thread per (graph, channel); 16 SSM states in registers; B/C
// broadcast through LDS each timestep.
// =====================================================================
__global__ __launch_bounds__(256) void scan_kernel(
    const float* __restrict__ xdbl, const float* __restrict__ delta,
    const float* __restrict__ xz,   const float* __restrict__ A_log,
    const float* __restrict__ Dp,   float* __restrict__ u)
{
    const int g = blockIdx.x >> 1;
    const int d = ((blockIdx.x & 1) << 8) + threadIdx.x;
    float hreg[DSTATE], Arow[DSTATE];
    #pragma unroll
    for (int n = 0; n < DSTATE; ++n) {
        Arow[n] = -__expf(A_log[d * DSTATE + n]);
        hreg[n] = 0.0f;
    }
    const float Dd = Dp[d];
    __shared__ float Bs[DSTATE], Cs[DSTATE];

    for (int l = 0; l < LSEQ; ++l) {
        size_t row = (size_t)g * LSEQ + l;
        if (threadIdx.x < 32) {
            float v = xdbl[row * DXDBL + DTRANK + threadIdx.x];
            if (threadIdx.x < DSTATE) Bs[threadIdx.x] = v;
            else                      Cs[threadIdx.x - DSTATE] = v;
        }
        __syncthreads();
        float dt = delta[row * DINNER + d];
        float uu = u[row * DINNER + d];
        float du = dt * uu;
        float y  = uu * Dd;
        #pragma unroll
        for (int n = 0; n < DSTATE; ++n) {
            float dA = __expf(dt * Arow[n]);
            hreg[n] = dA * hreg[n] + du * Bs[n];
            y += hreg[n] * Cs[n];
        }
        float z = xz[row * (2 * DINNER) + DINNER + d];
        y *= z / (1.0f + __expf(-z));                  // * silu(z)
        __syncthreads();                               // Bs/Cs reuse guard
        u[row * DINNER + d] = y;
    }
}

// =====================================================================
// host-side launcher
// =====================================================================
extern "C" void kernel_launch(void* const* d_in, const int* in_sizes, int n_in,
                              void* d_out, int out_size, void* d_ws, size_t ws_size,
                              hipStream_t stream)
{
    const float* x          = (const float*)d_in[0];
    const int*   edge_index = (const int*)  d_in[1];   // row 0 = sources
    const float* W_in       = (const float*)d_in[3];
    const float* b_in       = (const float*)d_in[4];
    const float* W_out      = (const float*)d_in[5];
    const float* b_out      = (const float*)d_in[6];
    const float* in_proj_w  = (const float*)d_in[7];
    const float* conv_w     = (const float*)d_in[8];
    const float* conv_b     = (const float*)d_in[9];
    const float* x_proj_w   = (const float*)d_in[10];
    const float* dt_proj_w  = (const float*)d_in[11];
    const float* dt_proj_b  = (const float*)d_in[12];
    const float* A_log      = (const float*)d_in[13];
    const float* D_param    = (const float*)d_in[14];
    const float* out_proj_w = (const float*)d_in[15];
    float* out = (float*)d_out;

    // workspace carve-up (fp32 unless noted); yout/scat reuse h/seq lifetimes
    char* ws = (char*)d_ws;
    size_t off = 0;
    auto take = [&](size_t bytes) { char* p = ws + off; off += (bytes + 255) & ~size_t(255); return p; };
    float*    h     = (float*)   take((size_t)NTOT * DMODEL * 4);      // 32 MB (reused as yout)
    float*    seq   = (float*)   take((size_t)NTOT * DMODEL * 4);      // 32 MB (reused as scat)
    float*    xz    = (float*)   take((size_t)NTOT * 2 * DINNER * 4);  // 128 MB
    float*    u     = (float*)   take((size_t)NTOT * DINNER * 4);      // 64 MB (becomes gated y)
    float*    delta = (float*)   take((size_t)NTOT * DINNER * 4);      // 64 MB
    float*    xdbl  = (float*)   take((size_t)NTOT * DXDBL * 4);       // 6 MB
    unsigned* deg   = (unsigned*)take((size_t)NTOT * 4);
    int*      order = (int*)     take((size_t)NTOT * 4);
    float* yout = h;     // free after gather
    float* scat = seq;   // free after xz GEMM

    const dim3 blk256(256);

    // 1) h = x @ W_in^T + b_in            [N,128]x[256,128]^T
    wmma_gemm_kernel<<<dim3(DMODEL / 64, NTOT / 128), blk256, 0, stream>>>(
        x, W_in, b_in, h, NTOT, DMODEL, DIN);

    // 2) degree of source nodes
    zero_u32_kernel<<<(NTOT + 255) / 256, blk256, 0, stream>>>(deg, NTOT);
    degree_kernel<<<(EEDGE + 255) / 256, blk256, 0, stream>>>(edge_index, deg, EEDGE);

    // 3) per-graph stable descending argsort
    sort_kernel<<<NG, 1024, 0, stream>>>(deg, order);

    // 4) seq = h[order]
    gather_kernel<<<(NTOT * 64 + 255) / 256, blk256, 0, stream>>>(h, order, seq);

    // 5) xz = seq @ in_proj_w^T           [N,256]x[1024,256]^T
    wmma_gemm_kernel<<<dim3(2 * DINNER / 64, NTOT / 128), blk256, 0, stream>>>(
        seq, in_proj_w, nullptr, xz, NTOT, 2 * DINNER, DMODEL);

    // 6) u = silu(causal_dwconv(xi) + b)
    conv_silu_kernel<<<(NTOT * DINNER + 255) / 256, blk256, 0, stream>>>(xz, conv_w, conv_b, u);

    // 7) x_dbl = u @ x_proj_w^T           [N,512]x[48,512]^T
    xproj_kernel<<<(NTOT * DXDBL + 255) / 256, blk256, 0, stream>>>(u, x_proj_w, xdbl);

    // 8) delta = softplus(dt @ dt_proj^T + b)
    delta_kernel<<<(NTOT * DINNER + 255) / 256, blk256, 0, stream>>>(xdbl, dt_proj_w, dt_proj_b, delta);

    // 9) selective scan + silu(z) gating -> u (in place)
    scan_kernel<<<NG * 2, blk256, 0, stream>>>(xdbl, delta, xz, A_log, D_param, u);

    // 10) yout = y @ out_proj_w^T         [N,512]x[256,512]^T
    wmma_gemm_kernel<<<dim3(DMODEL / 64, NTOT / 128), blk256, 0, stream>>>(
        u, out_proj_w, nullptr, yout, NTOT, DMODEL, DINNER);

    // 11) scatter back to node order, then final projection with bias
    scatter_kernel<<<(NTOT * 64 + 255) / 256, blk256, 0, stream>>>(yout, order, scat);
    wmma_gemm_kernel<<<dim3(DIN / 64, NTOT / 128), blk256, 0, stream>>>(
        scat, W_out, b_out, out, NTOT, DIN, DMODEL);
}
